// GAT_85572928405773
// MI455X (gfx1250) — compile-verified, matrix operand order checked by reading, not run on hardware
//
#include <hip/hip_runtime.h>

// ---------------- problem constants (match reference setup_inputs) ----------
#define NNODES 20000
#define NEDGES 320000
#define FIN    256
#define EDIM   16
#define NHEAD  8
#define CHEAD  64
#define HC     512     // NHEAD*CHEAD
#define NGRAPH 64
#define NCLS   10

// ---------------- WMMA types ------------------------------------------------
typedef __attribute__((ext_vector_type(16))) __bf16          v16bf;
typedef __attribute__((ext_vector_type(16))) unsigned short  v16u;
typedef __attribute__((ext_vector_type(8)))  float           v8f;

union FragU { v16u u; v16bf b; uint4 q[2]; };

__device__ __forceinline__ unsigned short f2bf(float f) {
    unsigned u = __float_as_uint(f);
    u += 0x7FFFu + ((u >> 16) & 1u);      // round-to-nearest-even
    return (unsigned short)(u >> 16);
}

// ---------------- GEMM: C[M,N] = A[M,K] @ B[K,N]  (f32 in, bf16 WMMA, f32 out)
// 256 threads = 8 waves arranged 4(M) x 2(N); block tile 64x128; wave tile
// 16x64 (4 WMMA accumulators); KC=64 per barrier round => 8 WMMA / round.
// LDS is staged "fragment-ready": As row-major over M (K contiguous),
// Bt transposed (row = output column, K contiguous); row stride 72 ushorts
// (144 B) keeps every lane's 16-B fragment halves 16-B aligned and the
// 16-lane ds_load_b128 gather bank-conflict-free (r*36 mod 64 distinct).
__global__ __launch_bounds__(256)
void gemm_wmma_bf16(const float* __restrict__ A, const float* __restrict__ B,
                    float* __restrict__ C, int M, int K, int N)
{
    __shared__ unsigned short As[64][72];    // M rows, k 0..63
    __shared__ unsigned short Bt[128][72];   // N cols, k 0..63

    const int tid  = threadIdx.x;
    const int lane = tid & 31;
    const int wid  = tid >> 5;       // 0..7
    const int wm   = wid >> 1;       // 0..3  (M sub-tile, 16 rows)
    const int wn   = wid & 1;        // 0..1  (N sub-tile, 64 cols)
    const int half = lane >> 4;      // 0..1
    const int mrow = lane & 15;      // 0..15
    const int bm   = blockIdx.x * 64;
    const int bn   = blockIdx.y * 128;

    v8f acc0 = {}, acc1 = {}, acc2 = {}, acc3 = {};

    for (int k0 = 0; k0 < K; k0 += 64) {
        if (k0 + 64 < K) {   // pull next tiles toward L2 (global_prefetch_b8)
            __builtin_prefetch(A + (size_t)(bm + wid * 8) * K + (k0 + 64), 0, 1);
            __builtin_prefetch(B + (size_t)(k0 + 64 + wid * 8) * N + bn, 0, 1);
        }
        // ---- stage A tile 64x64 (1024 float4 => 4 per thread) ----
        #pragma unroll
        for (int i = 0; i < 4; ++i) {
            int v  = tid + i * 256;          // 0..1023
            int r  = v >> 4;                 // M row 0..63
            int c4 = (v & 15) * 4;           // k 0,4,..,60
            int gr = bm + r;
            float4 f = make_float4(0.f, 0.f, 0.f, 0.f);
            if (gr < M && k0 + c4 + 3 < K)
                f = *(const float4*)&A[(size_t)gr * K + k0 + c4];
            uint2 pk;
            pk.x = (unsigned)f2bf(f.x) | ((unsigned)f2bf(f.y) << 16);
            pk.y = (unsigned)f2bf(f.z) | ((unsigned)f2bf(f.w) << 16);
            *(uint2*)&As[r][c4] = pk;        // ds_store_b64
        }
        // ---- stage B tile 64x128 transposed (2048 float4 => 8 per thread) --
        #pragma unroll
        for (int i = 0; i < 8; ++i) {
            int v  = tid + i * 256;          // 0..2047
            int r  = v >> 5;                 // k row 0..63
            int c4 = (v & 31) * 4;           // col 0,4,..,124
            int gc = bn + c4;
            float4 f = make_float4(0.f, 0.f, 0.f, 0.f);
            if (gc + 3 < N && k0 + r < K)
                f = *(const float4*)&B[(size_t)(k0 + r) * N + gc];
            Bt[c4 + 0][r] = f2bf(f.x);
            Bt[c4 + 1][r] = f2bf(f.y);
            Bt[c4 + 2][r] = f2bf(f.z);
            Bt[c4 + 3][r] = f2bf(f.w);
        }
        __syncthreads();

        const int arow = wm * 16 + mrow;
        const int bc0  = wn * 64 + mrow;
        #pragma unroll
        for (int kk = 0; kk < 2; ++kk) {
            const int ko = kk * 32;
            // A fragment: two contiguous 16-B halves (ds_load_b128 x2)
            FragU fa;
            fa.q[0] = *(const uint4*)&As[arow][ko + half * 8];
            fa.q[1] = *(const uint4*)&As[arow][ko + 16 + half * 8];
            // load all 4 B fragments first, then burst 4 WMMAs
            FragU fb0, fb1, fb2, fb3;
            fb0.q[0] = *(const uint4*)&Bt[bc0 +  0][ko + half * 8];
            fb0.q[1] = *(const uint4*)&Bt[bc0 +  0][ko + 16 + half * 8];
            fb1.q[0] = *(const uint4*)&Bt[bc0 + 16][ko + half * 8];
            fb1.q[1] = *(const uint4*)&Bt[bc0 + 16][ko + 16 + half * 8];
            fb2.q[0] = *(const uint4*)&Bt[bc0 + 32][ko + half * 8];
            fb2.q[1] = *(const uint4*)&Bt[bc0 + 32][ko + 16 + half * 8];
            fb3.q[0] = *(const uint4*)&Bt[bc0 + 48][ko + half * 8];
            fb3.q[1] = *(const uint4*)&Bt[bc0 + 48][ko + 16 + half * 8];
            acc0 = __builtin_amdgcn_wmma_f32_16x16x32_bf16(
                       false, fa.b, false, fb0.b, (short)0, acc0, false, false);
            acc1 = __builtin_amdgcn_wmma_f32_16x16x32_bf16(
                       false, fa.b, false, fb1.b, (short)0, acc1, false, false);
            acc2 = __builtin_amdgcn_wmma_f32_16x16x32_bf16(
                       false, fa.b, false, fb2.b, (short)0, acc2, false, false);
            acc3 = __builtin_amdgcn_wmma_f32_16x16x32_bf16(
                       false, fa.b, false, fb3.b, (short)0, acc3, false, false);
        }
        __syncthreads();
    }

    // ---- store: C/D layout VGPR r -> M = r + 8*half, N = lane&15 ----
    const int rowb = bm + wm * 16 + half * 8;
    #pragma unroll
    for (int r = 0; r < 8; ++r) {
        int row = rowb + r;
        if (row >= M) break;
        int c0 = bn + wn * 64 + mrow;
        if (c0 + 0  < N) C[(size_t)row * N + c0 + 0 ] = acc0[r];
        if (c0 + 16 < N) C[(size_t)row * N + c0 + 16] = acc1[r];
        if (c0 + 32 < N) C[(size_t)row * N + c0 + 32] = acc2[r];
        if (c0 + 48 < N) C[(size_t)row * N + c0 + 48] = acc3[r];
    }
}

// ---------------- attention scalars: a_s[n,h], a_d[n,h] ---------------------
__global__ void att_scores(const float* __restrict__ h,
                           const float* __restrict__ att_s,
                           const float* __restrict__ att_d,
                           float* __restrict__ as_, float* __restrict__ ad_)
{
    const int lane = threadIdx.x & 31;
    const int wid  = threadIdx.x >> 5;
    const int wpb  = blockDim.x >> 5;
    for (int p = blockIdx.x * wpb + wid; p < NNODES * NHEAD; p += gridDim.x * wpb) {
        int node = p >> 3, hh = p & 7;
        const float* hp = h + (size_t)node * HC + hh * CHEAD;
        const float* sp = att_s + hh * CHEAD;
        const float* dp = att_d + hh * CHEAD;
        float s = hp[lane] * sp[lane] + hp[lane + 32] * sp[lane + 32];
        float d = hp[lane] * dp[lane] + hp[lane + 32] * dp[lane + 32];
        #pragma unroll
        for (int off = 16; off >= 1; off >>= 1) {
            s += __shfl_xor(s, off, 32);
            d += __shfl_xor(d, off, 32);
        }
        if (lane == 0) { as_[p] = s; ad_[p] = d; }
    }
}

// ---------------- ve[d,h] = sum_c We[d, h*C+c] * att_e[h,c] -----------------
__global__ void compute_ve(const float* __restrict__ We,
                           const float* __restrict__ att_e,
                           float* __restrict__ ve)
{
    int t = threadIdx.x;                   // <<<1,128>>>
    if (t < EDIM * NHEAD) {
        int d = t >> 3, hh = t & 7;
        float s = 0.0f;
        for (int c = 0; c < CHEAD; ++c)
            s += We[(size_t)d * HC + hh * CHEAD + c] * att_e[hh * CHEAD + c];
        ve[t] = s;
    }
}

// ---------------- a_e[e,h] = edge_attr[e,:] @ ve ----------------------------
__global__ void edge_ae(const float* __restrict__ ea, const float* __restrict__ ve,
                        float* __restrict__ ae)
{
    __shared__ float sve[EDIM * NHEAD];
    if (threadIdx.x < EDIM * NHEAD) sve[threadIdx.x] = ve[threadIdx.x];
    __syncthreads();
    for (int e = blockIdx.x * blockDim.x + threadIdx.x; e < NEDGES;
         e += gridDim.x * blockDim.x) {
        const float4* eav = (const float4*)(ea + (size_t)e * EDIM);
        float4 q0 = eav[0], q1 = eav[1], q2 = eav[2], q3 = eav[3];
        float a[EDIM] = { q0.x, q0.y, q0.z, q0.w, q1.x, q1.y, q1.z, q1.w,
                          q2.x, q2.y, q2.z, q2.w, q3.x, q3.y, q3.z, q3.w };
        #pragma unroll
        for (int hh = 0; hh < NHEAD; ++hh) {
            float s = 0.0f;
            #pragma unroll
            for (int d = 0; d < EDIM; ++d) s += a[d] * sve[d * NHEAD + hh];
            ae[(size_t)e * NHEAD + hh] = s;
        }
    }
}

// ---------------- order-preserving float<->uint for atomic max --------------
__device__ __forceinline__ unsigned enc_f(float f) {
    unsigned u = __float_as_uint(f);
    return (u & 0x80000000u) ? ~u : (u | 0x80000000u);
}

__global__ void init_m(unsigned* __restrict__ m, int n) {
    int i = blockIdx.x * blockDim.x + threadIdx.x;
    if (i < n) m[i] = 0x007FFFFFu;        // enc(-inf)
}

// ---------------- alpha = leaky(a_s[src]+a_d[dst]+a_e, .2); segment max -----
__global__ void edge_logits(const int* __restrict__ src, const int* __restrict__ dst,
                            const float* __restrict__ as_, const float* __restrict__ ad_,
                            const float* __restrict__ ae, float* __restrict__ alpha,
                            unsigned* __restrict__ menc)
{
    const int total = NEDGES * NHEAD;
    for (int t = blockIdx.x * blockDim.x + threadIdx.x; t < total;
         t += gridDim.x * blockDim.x) {
        int e = t >> 3, hh = t & 7;
        int s = src[e], d = dst[e];
        float a = as_[s * NHEAD + hh] + ad_[d * NHEAD + hh] + ae[t];
        a = (a > 0.0f) ? a : 0.2f * a;
        alpha[t] = a;
        atomicMax(&menc[d * NHEAD + hh], enc_f(a));
    }
}

__global__ void decode_m(const unsigned* __restrict__ menc, float* __restrict__ m, int n) {
    int i = blockIdx.x * blockDim.x + threadIdx.x;
    if (i < n) {
        unsigned e = menc[i];
        unsigned u = (e & 0x80000000u) ? (e & 0x7FFFFFFFu) : ~e;
        float f = __uint_as_float(u);
        m[i] = (f > -3.0e38f && f < 3.0e38f) ? f : 0.0f;   // isfinite -> else 0
    }
}

// ---------------- ex = exp(alpha - m[dst]); denom[dst] += ex ----------------
__global__ void edge_exp(const int* __restrict__ dst, const float* __restrict__ m,
                         float* __restrict__ alpha_ex, float* __restrict__ denom)
{
    const int total = NEDGES * NHEAD;
    for (int t = blockIdx.x * blockDim.x + threadIdx.x; t < total;
         t += gridDim.x * blockDim.x) {
        int e = t >> 3, hh = t & 7;
        int d = dst[e];
        float ex = __expf(alpha_ex[t] - m[d * NHEAD + hh]);
        alpha_ex[t] = ex;
        atomicAdd(&denom[d * NHEAD + hh], ex);
    }
}

// ---------------- out[dst] += h[src] * (ex/denom[dst]) ; wave per edge ------
__global__ void edge_aggregate(const int* __restrict__ src, const int* __restrict__ dst,
                               const float* __restrict__ h, const float* __restrict__ ex,
                               const float* __restrict__ denom, float* __restrict__ out)
{
    const int lane = threadIdx.x & 31;
    const int wid  = threadIdx.x >> 5;
    const int wpb  = blockDim.x >> 5;
    for (int e = blockIdx.x * wpb + wid; e < NEDGES; e += gridDim.x * wpb) {
        int s = src[e], d = dst[e];
        const float* hs = h + (size_t)s * HC;
        float* od = out + (size_t)d * HC;
        #pragma unroll
        for (int hh = 0; hh < NHEAD; ++hh) {
            float w = ex[(size_t)e * NHEAD + hh] / (denom[d * NHEAD + hh] + 1e-16f);
            atomicAdd(&od[hh * CHEAD + lane],      hs[hh * CHEAD + lane] * w);
            atomicAdd(&od[hh * CHEAD + lane + 32], hs[hh * CHEAD + lane + 32] * w);
        }
    }
}

// ---------------- x = leaky(x + b, slope) -----------------------------------
__global__ void bias_act(float* __restrict__ x, const float* __restrict__ b, float slope)
{
    const int total = NNODES * HC;
    for (int i = blockIdx.x * blockDim.x + threadIdx.x; i < total;
         i += gridDim.x * blockDim.x) {
        float v = x[i] + b[i & (HC - 1)];
        x[i] = (v > 0.0f) ? v : slope * v;
    }
}

// ---------------- global mean pool: sums + counts ---------------------------
__global__ void pool_sum(const float* __restrict__ h, const int* __restrict__ batch,
                         float* __restrict__ pooled, float* __restrict__ cnt)
{
    const int total = NNODES * HC;
    for (int i = blockIdx.x * blockDim.x + threadIdx.x; i < total;
         i += gridDim.x * blockDim.x) {
        int n = i >> 9, c = i & (HC - 1);
        int g = batch[n];
        atomicAdd(&pooled[(size_t)g * HC + c], h[i]);
        if (c == 0) atomicAdd(&cnt[g], 1.0f);
    }
}

// ---------------- logits = (pooled/cnt) @ Wlin + blin -----------------------
__global__ void logits_k(const float* __restrict__ pooled, const float* __restrict__ cnt,
                         const float* __restrict__ Wlin, const float* __restrict__ blin,
                         float* __restrict__ logits)
{
    int t = blockIdx.x * blockDim.x + threadIdx.x;
    if (t < NGRAPH * NCLS) {
        int g = t / NCLS, cls = t % NCLS;
        float inv = 1.0f / fmaxf(cnt[g], 1.0f);
        float s = blin[cls];
        for (int c = 0; c < HC; ++c)
            s += pooled[(size_t)g * HC + c] * inv * Wlin[(size_t)c * NCLS + cls];
        logits[t] = s;
    }
}

__global__ void log_softmax_k(const float* __restrict__ logits, float* __restrict__ out)
{
    int g = blockIdx.x * blockDim.x + threadIdx.x;
    if (g < NGRAPH) {
        float mx = -1e30f;
        for (int c = 0; c < NCLS; ++c) mx = fmaxf(mx, logits[g * NCLS + c]);
        float se = 0.0f;
        for (int c = 0; c < NCLS; ++c) se += __expf(logits[g * NCLS + c] - mx);
        float lse = mx + __logf(se);
        for (int c = 0; c < NCLS; ++c) out[g * NCLS + c] = logits[g * NCLS + c] - lse;
    }
}

// ---------------- host-side layer driver ------------------------------------
static void run_gat_layer(const float* xin, int K,
                          const float* W, const float* atts, const float* attd,
                          const float* We, const float* atte, const float* b,
                          const int* src, const int* dst, const float* ea,
                          float* bufH, float* bufO,
                          float* as_b, float* ad_b, unsigned* menc, float* mf,
                          float* denom, float* aeb, float* exb, float* veb,
                          hipStream_t stream)
{
    dim3 gg((NNODES + 63) / 64, (HC + 127) / 128);
    gemm_wmma_bf16<<<gg, 256, 0, stream>>>(xin, W, bufH, NNODES, K, HC);
    att_scores<<<2048, 256, 0, stream>>>(bufH, atts, attd, as_b, ad_b);
    compute_ve<<<1, 128, 0, stream>>>(We, atte, veb);
    edge_ae<<<2048, 256, 0, stream>>>(ea, veb, aeb);
    init_m<<<(NNODES * NHEAD + 255) / 256, 256, 0, stream>>>(menc, NNODES * NHEAD);
    edge_logits<<<4096, 256, 0, stream>>>(src, dst, as_b, ad_b, aeb, exb, menc);
    decode_m<<<(NNODES * NHEAD + 255) / 256, 256, 0, stream>>>(menc, mf, NNODES * NHEAD);
    hipMemsetAsync(denom, 0, (size_t)NNODES * NHEAD * sizeof(float), stream);
    edge_exp<<<4096, 256, 0, stream>>>(dst, mf, exb, denom);
    hipMemsetAsync(bufO, 0, (size_t)NNODES * HC * sizeof(float), stream);
    edge_aggregate<<<8192, 256, 0, stream>>>(src, dst, bufH, exb, denom, bufO);
    bias_act<<<4096, 256, 0, stream>>>(bufO, b, 0.01f);
}

extern "C" void kernel_launch(void* const* d_in, const int* in_sizes, int n_in,
                              void* d_out, int out_size, void* d_ws, size_t ws_size,
                              hipStream_t stream)
{
    (void)in_sizes; (void)n_in; (void)out_size; (void)ws_size;
    const float* x     = (const float*)d_in[0];
    const int*   eidx  = (const int*)  d_in[1];
    const float* ea    = (const float*)d_in[2];
    const int*   batch = (const int*)  d_in[3];
    const float* W0    = (const float*)d_in[4];
    const float* asc0  = (const float*)d_in[5];
    const float* adc0  = (const float*)d_in[6];
    const float* We0   = (const float*)d_in[7];
    const float* aec0  = (const float*)d_in[8];
    const float* b0    = (const float*)d_in[9];
    const float* W1    = (const float*)d_in[10];
    const float* asc1  = (const float*)d_in[11];
    const float* adc1  = (const float*)d_in[12];
    const float* We1   = (const float*)d_in[13];
    const float* aec1  = (const float*)d_in[14];
    const float* b1    = (const float*)d_in[15];
    const float* Wlin  = (const float*)d_in[16];
    const float* blin  = (const float*)d_in[17];

    const int* src = eidx;
    const int* dst = eidx + NEDGES;

    // ------- carve workspace (~105 MB) -------
    char* p = (char*)d_ws;
    auto carve = [&](size_t bytes) -> char* {
        char* r = p; p += (bytes + 255) & ~(size_t)255; return r;
    };
    float*    bufH   = (float*)   carve((size_t)NNODES * HC * 4);    // transformed h
    float*    bufO   = (float*)   carve((size_t)NNODES * HC * 4);    // aggregated / hact
    float*    as_b   = (float*)   carve((size_t)NNODES * NHEAD * 4);
    float*    ad_b   = (float*)   carve((size_t)NNODES * NHEAD * 4);
    unsigned* menc   = (unsigned*)carve((size_t)NNODES * NHEAD * 4);
    float*    mf     = (float*)   carve((size_t)NNODES * NHEAD * 4);
    float*    denom  = (float*)   carve((size_t)NNODES * NHEAD * 4);
    float*    aeb    = (float*)   carve((size_t)NEDGES * NHEAD * 4);
    float*    exb    = (float*)   carve((size_t)NEDGES * NHEAD * 4);
    float*    veb    = (float*)   carve(EDIM * NHEAD * 4);
    float*    pooled = (float*)   carve((size_t)NGRAPH * HC * 4);
    float*    cntb   = (float*)   carve(NGRAPH * 4);
    float*    logit  = (float*)   carve(NGRAPH * NCLS * 4);

    // ------- layer 0: x[N,256] -> hact0 in bufO -------
    run_gat_layer(x, FIN, W0, asc0, adc0, We0, aec0, b0,
                  src, dst, ea, bufH, bufO,
                  as_b, ad_b, menc, mf, denom, aeb, exb, veb, stream);

    // ------- layer 1: hact0[N,512] -> hact1 in bufO (bufH reused for h1) ----
    run_gat_layer(bufO, HC, W1, asc1, adc1, We1, aec1, b1,
                  src, dst, ea, bufH, bufO,
                  as_b, ad_b, menc, mf, denom, aeb, exb, veb, stream);

    // ------- pool + classifier + log-softmax -------
    hipMemsetAsync(pooled, 0, (size_t)NGRAPH * HC * sizeof(float), stream);
    hipMemsetAsync(cntb, 0, NGRAPH * sizeof(float), stream);
    pool_sum<<<4096, 256, 0, stream>>>(bufO, batch, pooled, cntb);
    logits_k<<<(NGRAPH * NCLS + 255) / 256, 256, 0, stream>>>(pooled, cntb, Wlin, blin, logit);
    log_softmax_k<<<1, 64, 0, stream>>>(logit, (float*)d_out);
}
